// KNNLocalAttention_87282325389618
// MI455X (gfx1250) — matrix operand order repaired and checked by reading, not compiled, not run on hardware
//
#include <hip/hip_runtime.h>
#include <hip/hip_bf16.h>
#include <math.h>

// ---------------------------------------------------------------------------
// KNN local attention for MI455X (gfx1250, wave32, WMMA).
//   1) knn_kernel        : top-16 neighbours (VALU, LDS tiles via async-to-LDS)
//   2) cvt kernels       : f32 -> f16 operand staging (weights transposed)
//   3) linear_f16_kernel : Y = Xh @ Wt^T + b, wave keeps 16x256 A strip in
//                          registers, 64 unrolled v_wmma_f32_16x16x32_f16/wave
//   4) attn_kernel       : fused pos-MLP (WMMA) + neighbour attention
//   5) ln_kernel         : mask, residual, LayerNorm epilogue
// ---------------------------------------------------------------------------

typedef __attribute__((ext_vector_type(16))) _Float16 v16h;
typedef __attribute__((ext_vector_type(8)))  _Float16 v8h;
typedef __attribute__((ext_vector_type(2)))  _Float16 v2h;
typedef __attribute__((ext_vector_type(8)))  float    v8f;

#define DD   256          // feature dim (compile-time for full unrolling)
#define HH   8            // heads
#define DH   32           // head dim
#define KN   16           // neighbours
#define TS   256          // knn tile size (= blockDim)
#define LN_EPS 1e-6f

#if defined(__has_builtin)
#if __has_builtin(__builtin_amdgcn_global_load_async_to_lds_b32) && \
    __has_builtin(__builtin_amdgcn_s_wait_asynccnt)
#define USE_ASYNC_LDS 1
#endif
#endif

#if USE_ASYNC_LDS
typedef __attribute__((address_space(1))) int gas_int;   // global
typedef __attribute__((address_space(3))) int las_int;   // LDS
#endif

// ---------------------------------------------------------------------------
// 1) kNN: one thread per query; sorted 16-entry best list in registers.
//    Coordinate tiles staged to LDS with CDNA5 async copies when available.
// ---------------------------------------------------------------------------
__global__ void knn_kernel(const float* __restrict__ coords,
                           int* __restrict__ idxout, int N)
{
    const int gq  = blockIdx.x * blockDim.x + threadIdx.x;
    const int b   = gq / N;
    const int n   = gq % N;
    const float* cb = coords + (size_t)b * N * 3;

    const float qx = cb[(size_t)n * 3 + 0];
    const float qy = cb[(size_t)n * 3 + 1];
    const float qz = cb[(size_t)n * 3 + 2];

    __shared__ float sx[TS], sy[TS], sz[TS];

    float bd[KN];
    int   bi[KN];
#pragma unroll
    for (int t = 0; t < KN; ++t) { bd[t] = INFINITY; bi[t] = 0; }

    for (int base = 0; base < N; base += TS) {
        __syncthreads();
        const int j = base + threadIdx.x;            // N % TS == 0
#if USE_ASYNC_LDS
        __builtin_amdgcn_global_load_async_to_lds_b32(
            (gas_int*)(cb + (size_t)j * 3 + 0),
            (las_int*)&sx[threadIdx.x], 0, 0);
        __builtin_amdgcn_global_load_async_to_lds_b32(
            (gas_int*)(cb + (size_t)j * 3 + 1),
            (las_int*)&sy[threadIdx.x], 0, 0);
        __builtin_amdgcn_global_load_async_to_lds_b32(
            (gas_int*)(cb + (size_t)j * 3 + 2),
            (las_int*)&sz[threadIdx.x], 0, 0);
        __builtin_amdgcn_s_wait_asynccnt(0);
#else
        sx[threadIdx.x] = cb[(size_t)j * 3 + 0];
        sy[threadIdx.x] = cb[(size_t)j * 3 + 1];
        sz[threadIdx.x] = cb[(size_t)j * 3 + 2];
#endif
        __syncthreads();

        for (int t = 0; t < TS; ++t) {
            const float dx = qx - sx[t];
            const float dy = qy - sy[t];
            const float dz = qz - sz[t];
            const float d2 = dx * dx + dy * dy + dz * dz;
            if (d2 < bd[KN - 1]) {
                bd[KN - 1] = d2;
                bi[KN - 1] = base + t;
#pragma unroll
                for (int s = KN - 1; s > 0; --s) {
                    if (bd[s] < bd[s - 1]) {
                        float td = bd[s]; bd[s] = bd[s - 1]; bd[s - 1] = td;
                        int   ti = bi[s]; bi[s] = bi[s - 1]; bi[s - 1] = ti;
                    }
                }
            }
        }
    }
#pragma unroll
    for (int t = 0; t < KN; ++t)
        idxout[(size_t)gq * KN + t] = bi[t];
}

// ---------------------------------------------------------------------------
// 2a) elementwise f32 -> f16 (pairs, emits v_cvt_pk_f16_f32)
// ---------------------------------------------------------------------------
__global__ void cvt_f16_kernel(const float* __restrict__ src,
                               _Float16* __restrict__ dst, int npair)
{
    const int i = blockIdx.x * blockDim.x + threadIdx.x;
    if (i < npair) {
        const float2 v = ((const float2*)src)[i];
        v2h o; o[0] = (_Float16)v.x; o[1] = (_Float16)v.y;
        ((v2h*)dst)[i] = o;
    }
}

// 2b) weight transpose + convert: Wt[n][k] = (f16) W[k][n]   (256 x 256)
__global__ void cvt_w_kernel(const float* __restrict__ W,
                             _Float16* __restrict__ Wt)
{
    const int n = blockIdx.x;        // 256 blocks
    const int k = threadIdx.x;       // 256 threads
    Wt[(size_t)n * DD + k] = (_Float16)W[(size_t)k * DD + n];
}

// ---------------------------------------------------------------------------
// WMMA fragment helpers (wave32 16-bit layouts, ISA 7.12.2)
// A (16x32): lane row = lane&15; contiguous k run k0+16*half .. +16.
// B (32x16): lane col = lane&15; contiguous k run k0+16*half .. +16 in Wt[n][k].
// ---------------------------------------------------------------------------
__device__ __forceinline__ v16h load_a_frag(const _Float16* rowp, int k0, int half)
{
    return *(const v16h*)(rowp + k0 + half * 16);
}

// ---------------------------------------------------------------------------
// 3) Linear: Y[M,256] = Xh[M,256] @ Wt[n][k]^T + bias.
//    Wave owns a 16-row strip: full A strip (16x256 f16) in registers,
//    sweeps 8 column tiles -> 64 WMMAs per wave, K-loop fully unrolled.
// ---------------------------------------------------------------------------
__global__ void linear_f16_kernel(const _Float16* __restrict__ Xh,
                                  const _Float16* __restrict__ Wt,
                                  const float* __restrict__ bias,
                                  float* __restrict__ Y, int M)
{
    const int lane = threadIdx.x & 31;
    const int wave = threadIdx.x >> 5;
    const int task = blockIdx.x * 8 + wave;          // wave-uniform
    const int nStrips = M >> 4;
    if (task >= nStrips * 2) return;                 // wave-uniform exit

    const int strip = task >> 1;                     // 16-row strip
    const int nhalf = task & 1;                      // 8 column tiles each
    const int l15   = lane & 15;
    const int half  = lane >> 4;
    const int m     = strip * 16 + l15;

    // A strip: 8 ktiles, 64 VGPRs of f16
    v16h a[8];
    const _Float16* xrow = Xh + (size_t)m * DD;
#pragma unroll
    for (int kt = 0; kt < 8; ++kt)
        a[kt] = load_a_frag(xrow, kt * 32, half);

    for (int tn = nhalf * 8; tn < nhalf * 8 + 8; ++tn) {
        const int nc = tn * 16 + l15;
        const _Float16* wcol = Wt + (size_t)nc * DD;
        if (tn + 1 < nhalf * 8 + 8)
            __builtin_prefetch(wcol + 16 * DD, 0, 1);

        v8f acc = {};
#pragma unroll
        for (int kt = 0; kt < 8; ++kt) {
            const v16h bf = *(const v16h*)(wcol + kt * 32 + half * 16);
            acc = __builtin_amdgcn_wmma_f32_16x16x32_f16(
                false, a[kt], false, bf, (short)0, acc, false, false);
        }
        const float bb = bias[nc];
#pragma unroll
        for (int r = 0; r < 8; ++r) {
            const int mm = strip * 16 + r + 8 * half;
            Y[(size_t)mm * DD + nc] = acc[r] + bb;
        }
    }
}

// ---------------------------------------------------------------------------
// 4) Fused pos-MLP + neighbour attention. One block (8 waves) per query.
// ---------------------------------------------------------------------------
__global__ void attn_kernel(const float* __restrict__ coords,
                            const float* __restrict__ normals,
                            const float* __restrict__ mask,
                            const int*   __restrict__ idx,
                            const float* __restrict__ qg,
                            const float* __restrict__ kg,
                            const float* __restrict__ vg,
                            const float* __restrict__ Wp1,
                            const float* __restrict__ bp1,
                            const _Float16* __restrict__ Wp2t, // [n][k] f16
                            const float* __restrict__ bp2,
                            float* __restrict__ attn_out,
                            int N)
{
    const int n   = blockIdx.x % N;
    const int b   = blockIdx.x / N;
    const int row = b * N + n;
    const int tid = threadIdx.x;

    __shared__ int      s_idx[KN];
    __shared__ float    s_pin[KN][8];        // rel_pos(3) + normals(3)
    __shared__ _Float16 s_hid[KN][DD];       // pos-MLP hidden (post-GELU)
    __shared__ float    s_pe [KN][DD];       // pos_emb
    __shared__ float    s_log[KN][HH];
    __shared__ float    s_att[KN][HH];

    if (tid < KN) {
        const int j = idx[(size_t)row * KN + tid];
        s_idx[tid] = j;
        const float* cq = coords + (size_t)row * 3;
        const float* cj = coords + ((size_t)b * N + j) * 3;
        const float* nr = normals + (size_t)row * 3;
        s_pin[tid][0] = cq[0] - cj[0];
        s_pin[tid][1] = cq[1] - cj[1];
        s_pin[tid][2] = cq[2] - cj[2];
        s_pin[tid][3] = nr[0];
        s_pin[tid][4] = nr[1];
        s_pin[tid][5] = nr[2];
    }
    __syncthreads();

    // --- pos-MLP layer 1 + exact GELU (K=6 contraction, VALU) --------------
    {
        const int c = tid;
        const float w0 = Wp1[0 * DD + c], w1 = Wp1[1 * DD + c],
                    w2 = Wp1[2 * DD + c], w3 = Wp1[3 * DD + c],
                    w4 = Wp1[4 * DD + c], w5 = Wp1[5 * DD + c];
        const float bb = bp1[c];
#pragma unroll
        for (int kn = 0; kn < KN; ++kn) {
            const float x = s_pin[kn][0] * w0 + s_pin[kn][1] * w1 +
                            s_pin[kn][2] * w2 + s_pin[kn][3] * w3 +
                            s_pin[kn][4] * w4 + s_pin[kn][5] * w5 + bb;
            const float g = 0.5f * x * (1.0f + erff(x * 0.7071067811865475f));
            s_hid[kn][c] = (_Float16)g;
        }
    }
    __syncthreads();

    // --- pos-MLP layer 2: (16 x 256) @ (256 x 256) via WMMA ----------------
    {
        const int lane = tid & 31, wave = tid >> 5;
        const int l15 = lane & 15, half = lane >> 4;
        for (int t = 0; t < 2; ++t) {                 // 8 waves * 2 = 16 tiles
            const int tn = wave * 2 + t;
            const int nc = tn * 16 + l15;
            const _Float16* wcol = Wp2t + (size_t)nc * DD;
            v8f acc = {};
#pragma unroll
            for (int kt = 0; kt < 8; ++kt) {
                const v16h af = load_a_frag(&s_hid[l15][0], kt * 32, half);
                const v16h bf = *(const v16h*)(wcol + kt * 32 + half * 16);
                acc = __builtin_amdgcn_wmma_f32_16x16x32_f16(
                    false, af, false, bf, (short)0, acc, false, false);
            }
            const float bcol = bp2[nc];
#pragma unroll
            for (int r = 0; r < 8; ++r)
                s_pe[r + 8 * half][nc] = acc[r] + bcol;
        }
    }
    __syncthreads();

    // --- logits: q . (k_gather + pos_emb), per (neighbour, head) -----------
    const float scale = 0.17677669529663687f;          // 1/sqrt(32)
    if (tid < KN * HH) {
        const int kn = tid / HH;
        const int h  = tid % HH;
        const int j  = s_idx[kn];
        const float* krow = kg + ((size_t)b * N + j) * DD + h * DH;
        const float* qrow = qg + (size_t)row * DD + h * DH;
        const float* pe   = &s_pe[kn][h * DH];
        float acc = 0.f;
#pragma unroll
        for (int d = 0; d < DH; ++d)
            acc += qrow[d] * (krow[d] + pe[d]);
        const float valid = mask[(size_t)b * N + j];
        s_log[kn][h] = (valid > 0.f) ? acc * scale : -INFINITY;
    }
    __syncthreads();

    // --- softmax over neighbours (per head) --------------------------------
    if (tid < HH) {
        const int h = tid;
        float mx = -INFINITY;
#pragma unroll
        for (int kn = 0; kn < KN; ++kn) mx = fmaxf(mx, s_log[kn][h]);
        if (mx == -INFINITY) {                         // nan_to_num path
#pragma unroll
            for (int kn = 0; kn < KN; ++kn) s_att[kn][h] = 0.f;
        } else {
            float e[KN], sum = 0.f;
#pragma unroll
            for (int kn = 0; kn < KN; ++kn) {
                const float l = s_log[kn][h];
                e[kn] = (l == -INFINITY) ? 0.f : expf(l - mx);
                sum += e[kn];
            }
            const float inv = 1.0f / sum;
#pragma unroll
            for (int kn = 0; kn < KN; ++kn) s_att[kn][h] = e[kn] * inv;
        }
    }
    __syncthreads();

    // --- weighted sum of (v_gather + pos_emb) ------------------------------
    {
        const int c = tid;
        const int h = c >> 5;                          // c / DH
        float acc = 0.f;
#pragma unroll
        for (int kn = 0; kn < KN; ++kn) {
            const int j = s_idx[kn];
            acc += s_att[kn][h] * (vg[((size_t)b * N + j) * DD + c] + s_pe[kn][c]);
        }
        attn_out[(size_t)row * DD + c] = acc;
    }
}

// ---------------------------------------------------------------------------
// 5) mask * proj + residual, LayerNorm, gamma/beta, final mask.
// ---------------------------------------------------------------------------
__global__ void ln_kernel(const float* __restrict__ feat,
                          const float* __restrict__ proj,
                          const float* __restrict__ mask,
                          const float* __restrict__ gamma,
                          const float* __restrict__ beta,
                          float* __restrict__ out)
{
    const int row = blockIdx.x;
    const int c   = threadIdx.x;
    __shared__ float red[DD];

    const float m = mask[row];
    const float h = feat[(size_t)row * DD + c] + proj[(size_t)row * DD + c] * m;

    red[c] = h;
    __syncthreads();
    for (int s = DD / 2; s > 0; s >>= 1) {
        if (c < s) red[c] += red[c + s];
        __syncthreads();
    }
    const float mu = red[0] * (1.0f / DD);
    __syncthreads();

    const float d = h - mu;
    red[c] = d * d;
    __syncthreads();
    for (int s = DD / 2; s > 0; s >>= 1) {
        if (c < s) red[c] += red[c + s];
        __syncthreads();
    }
    const float var = red[0] * (1.0f / DD);

    const float y = d * rsqrtf(var + LN_EPS) * gamma[c] + beta[c];
    out[(size_t)row * DD + c] = y * m;
}

// ---------------------------------------------------------------------------
// Launch
// ---------------------------------------------------------------------------
extern "C" void kernel_launch(void* const* d_in, const int* in_sizes, int n_in,
                              void* d_out, int out_size, void* d_ws, size_t ws_size,
                              hipStream_t stream)
{
    const float* features = (const float*)d_in[0];
    const float* coords   = (const float*)d_in[1];
    const float* normals  = (const float*)d_in[2];
    const float* mask     = (const float*)d_in[3];
    const float* Wq = (const float*)d_in[4];   const float* bq = (const float*)d_in[5];
    const float* Wk = (const float*)d_in[6];   const float* bk = (const float*)d_in[7];
    const float* Wv = (const float*)d_in[8];   const float* bv = (const float*)d_in[9];
    const float* Wo = (const float*)d_in[10];  const float* bo = (const float*)d_in[11];
    const float* Wp1 = (const float*)d_in[12]; const float* bp1 = (const float*)d_in[13];
    const float* Wp2 = (const float*)d_in[14]; const float* bp2 = (const float*)d_in[15];
    const float* gamma = (const float*)d_in[16];
    const float* beta  = (const float*)d_in[17];

    const int D  = in_sizes[5];          // 256
    const int BN = in_sizes[0] / D;      // B*N
    const int B  = 2;                    // per reference setup
    const int N  = BN / B;               // 8192

    // Workspace carve-up (256B aligned)
    char*  ws    = (char*)d_ws;
    size_t off   = 0;
    auto   carve = [&](size_t bytes) {
        char* p = ws + off;
        off += (bytes + 255) & ~(size_t)255;
        return p;
    };
    int*      idx      = (int*)     carve((size_t)BN * KN * sizeof(int));
    float*    q        = (float*)   carve((size_t)BN * D * sizeof(float));
    float*    k        = (float*)   carve((size_t)BN * D * sizeof(float));
    float*    v        = (float*)   carve((size_t)BN * D * sizeof(float));
    float*    attn_out = (float*)   carve((size_t)BN * D * sizeof(float));
    float*    proj     = (float*)   carve((size_t)BN * D * sizeof(float));
    _Float16* Xh       = (_Float16*)carve((size_t)BN * D * sizeof(_Float16));
    _Float16* Ah       = (_Float16*)carve((size_t)BN * D * sizeof(_Float16));
    _Float16* Wqt      = (_Float16*)carve((size_t)D * D * sizeof(_Float16));
    _Float16* Wkt      = (_Float16*)carve((size_t)D * D * sizeof(_Float16));
    _Float16* Wvt      = (_Float16*)carve((size_t)D * D * sizeof(_Float16));
    _Float16* Wot      = (_Float16*)carve((size_t)D * D * sizeof(_Float16));
    _Float16* Wp2t     = (_Float16*)carve((size_t)D * D * sizeof(_Float16));
    (void)ws_size; (void)n_in; (void)out_size;

    // 1) kNN indices
    knn_kernel<<<BN / TS, TS, 0, stream>>>(coords, idx, N);

    // 2) operand staging: activations f16, weights f16 transposed
    const int npair = BN * D / 2;
    cvt_f16_kernel<<<(npair + 255) / 256, 256, 0, stream>>>(features, Xh, npair);
    cvt_w_kernel<<<D, D, 0, stream>>>(Wq,  Wqt);
    cvt_w_kernel<<<D, D, 0, stream>>>(Wk,  Wkt);
    cvt_w_kernel<<<D, D, 0, stream>>>(Wv,  Wvt);
    cvt_w_kernel<<<D, D, 0, stream>>>(Wo,  Wot);
    cvt_w_kernel<<<D, D, 0, stream>>>(Wp2, Wp2t);

    // 3) q/k/v projections (WMMA, 2 wave-tasks per 16-row strip)
    const int tasks    = (BN / 16) * 2;
    const int gemmBlks = (tasks + 7) / 8;
    linear_f16_kernel<<<gemmBlks, 256, 0, stream>>>(Xh, Wqt, bq, q, BN);
    linear_f16_kernel<<<gemmBlks, 256, 0, stream>>>(Xh, Wkt, bk, k, BN);
    linear_f16_kernel<<<gemmBlks, 256, 0, stream>>>(Xh, Wvt, bv, v, BN);

    // 4) fused pos-MLP + attention (WMMA inside)
    attn_kernel<<<BN, 256, 0, stream>>>(coords, normals, mask, idx, q, k, v,
                                        Wp1, bp1, Wp2t, bp2, attn_out, N);

    // 5) output projection (WMMA)
    cvt_f16_kernel<<<(npair + 255) / 256, 256, 0, stream>>>(attn_out, Ah, npair);
    linear_f16_kernel<<<gemmBlks, 256, 0, stream>>>(Ah, Wot, bo, proj, BN);

    // 6) residual + LayerNorm epilogue
    ln_kernel<<<BN, 256, 0, stream>>>(features, proj, mask, gamma, beta,
                                      (float*)d_out);
}
